// SelfAttentionDual_65481071397373
// MI455X (gfx1250) — matrix operand order repaired and checked by reading, not compiled, not run on hardware
//
#include <hip/hip_runtime.h>

// ---- problem constants (match reference) ----
#define B_ 4
#define S_ 2048
#define E_ 256
#define H_ 8

typedef __attribute__((ext_vector_type(16))) __bf16 v16bf;
typedef __attribute__((ext_vector_type(8)))  float  v8f;
typedef unsigned int u32;
typedef unsigned short u16;
typedef __attribute__((ext_vector_type(4))) u32 u32x4;

union BF16x16 { v16bf v; u32x4 u[2]; };

__device__ __forceinline__ u16 f2bf(float f) {
    u32 u = __float_as_uint(f);
    u += 0x7FFFu + ((u >> 16) & 1u);   // round-to-nearest-even
    return (u16)(u >> 16);
}

__device__ __forceinline__ v8f wmma_bf16(v16bf a, v16bf b, v8f c) {
    // D = A(16x32 bf16) * B(32x16 bf16) + C(16x16 f32)
    return __builtin_amdgcn_wmma_f32_16x16x32_bf16(
        false, a, false, b, (short)0, c, false, false);
}

// gfx1250 async global->LDS copy: each lane copies 16B, tracked by ASYNCcnt.
__device__ __forceinline__ void async_ld16(u32 lds_byte_off, const void* g) {
    asm volatile("global_load_async_to_lds_b128 %0, %1, off"
                 :: "v"(lds_byte_off), "v"((unsigned long long)(uintptr_t)g)
                 : "memory");
}
__device__ __forceinline__ void wait_async_le8() {
    asm volatile("s_wait_asynccnt 0x8" ::: "memory");
}
__device__ __forceinline__ void wait_async_0() {
    asm volatile("s_wait_asynccnt 0x0" ::: "memory");
}

// ------------------------------------------------------------------
// Kernel 1: fp32 -> bf16 elementwise convert
// ------------------------------------------------------------------
__global__ __launch_bounds__(256) void cvt_bf16_kernel(
    const float* __restrict__ src, u16* __restrict__ dst, int n) {
    int i = blockIdx.x * 256 + threadIdx.x;
    if (i < n) dst[i] = f2bf(src[i]);
}

// ------------------------------------------------------------------
// Kernel 2: fused Q/K/V projection (one wave = one 16x16 tile of all 3).
//   Q[b,h,s,f]  = (1/16)*sum_e x wq   (softmax scale folded in)
//   Kt[b,h,f,s] =        sum_e x wk   (stored transposed)
//   V[b,h,s,f]  =        sum_e x wv
// ------------------------------------------------------------------
__global__ __launch_bounds__(256) void proj_qkv_kernel(
    const u16* __restrict__ xb,
    const u16* __restrict__ wqb, const u16* __restrict__ wkb,
    const u16* __restrict__ wvb,
    u16* __restrict__ Qb, u16* __restrict__ Kt, u16* __restrict__ Vb)
{
    const int lane = threadIdx.x & 31;
    const int wave = threadIdx.x >> 5;
    int tile = blockIdx.x * 8 + wave;          // 65536 wave-tiles
    const int ft = tile & 15;  tile >>= 4;
    const int st = tile & 127; tile >>= 7;
    const int h  = tile & 7;   tile >>= 3;
    const int b  = tile;

    const int row = lane & 15;
    const int kh  = lane >> 4;

    v8f cq = {}, ck = {}, cv = {};
    const u16* xrow = xb + ((size_t)(b * S_ + st * 16 + row)) * E_;

    #pragma unroll
    for (int e0 = 0; e0 < E_; e0 += 32) {
        BF16x16 a;
        a.u[0] = *(const u32x4*)(xrow + e0 + kh * 8);
        a.u[1] = *(const u32x4*)(xrow + e0 + 16 + kh * 8);
        const size_t wb = ((size_t)h * E_ + e0) * E_ + ft * 16 + row;
        BF16x16 bq, bk, bv;
        #pragma unroll
        for (int s = 0; s < 16; s += 2) {
            const int k = (s & 7) + 8 * kh + (s & 8) * 2;
            const size_t o0 = wb + (size_t)k * E_, o1 = o0 + E_;
            bq.u[s >> 3][(s & 7) >> 1] = (u32)wqb[o0] | ((u32)wqb[o1] << 16);
            bk.u[s >> 3][(s & 7) >> 1] = (u32)wkb[o0] | ((u32)wkb[o1] << 16);
            bv.u[s >> 3][(s & 7) >> 1] = (u32)wvb[o0] | ((u32)wvb[o1] << 16);
        }
        cq = wmma_bf16(a.v, bq.v, cq);
        ck = wmma_bf16(a.v, bk.v, ck);
        cv = wmma_bf16(a.v, bv.v, cv);
    }

    const size_t bh = (size_t)(b * H_ + h);
    const int N = lane & 15;
    #pragma unroll
    for (int r = 0; r < 8; ++r) {
        const int M = r + 8 * kh;
        const size_t off = (bh * S_ + st * 16 + M) * E_ + ft * 16 + N;
        Qb[off] = f2bf(cq[r] * 0.0625f);
        Vb[off] = f2bf(cv[r]);
    }
    u32x4 pk;
    pk.x = (u32)f2bf(ck[0]) | ((u32)f2bf(ck[1]) << 16);
    pk.y = (u32)f2bf(ck[2]) | ((u32)f2bf(ck[3]) << 16);
    pk.z = (u32)f2bf(ck[4]) | ((u32)f2bf(ck[5]) << 16);
    pk.w = (u32)f2bf(ck[6]) | ((u32)f2bf(ck[7]) << 16);
    const size_t koff = (bh * E_ + ft * 16 + N) * S_ + st * 16 + kh * 8;
    *(u32x4*)(Kt + koff) = pk;
}

// ------------------------------------------------------------------
// Kernel 3: flash attention. 8 waves/WG; wave owns 16 q-rows x E=256.
// K/V tiles (32 kv rows) staged in LDS via async copies, double-buffered.
// LDS row pitches padded (K: 80B, V: 528B, P: 80B) to spread bank groups.
// ------------------------------------------------------------------
#define KPITCH 40     // u16 per K-tile LDS row  (32 data + 8 pad)
#define VPITCH 264    // u16 per V-tile LDS row  (256 data + 8 pad)
#define PPITCH 40     // u16 per P-tile LDS row

__global__ __launch_bounds__(256) void attn_kernel(
    const u16* __restrict__ Qb, const u16* __restrict__ Kt,
    const u16* __restrict__ Vb, float* __restrict__ Obuf)
{
    __shared__ u16 kbuf[2][256 * KPITCH];   // 2 x 20 KB
    __shared__ u16 vbuf[2][32 * VPITCH];    // 2 x 16.5 KB
    __shared__ u16 plds[8 * 16 * PPITCH];   // 10 KB (per-wave P staging)

    const int lane = threadIdx.x & 31;
    const int wave = threadIdx.x >> 5;
    int id = blockIdx.x;                    // B*H*(S/128) = 512
    const int qt = id & 15; id >>= 4;
    const int h  = id & 7;  id >>= 3;
    const int b  = id;
    const size_t bh = (size_t)(b * H_ + h);

    const int row = lane & 15;
    const int kh  = lane >> 4;
    const int q0  = qt * 128 + wave * 16;

    const u16* qrow = Qb + (bh * S_ + q0 + row) * E_;   // hot in WGP$
    const u16* ktb  = Kt + bh * (size_t)E_ * S_;
    const u16* vbp  = Vb + bh * (size_t)S_ * E_;

    float m[8], l[8];
    v8f o[16];
    {
        v8f z = {};
        #pragma unroll
        for (int t = 0; t < 16; ++t) o[t] = z;
        #pragma unroll
        for (int r = 0; r < 8; ++r) { m[r] = -1e30f; l[r] = 0.0f; }
    }
    u16* pw = plds + wave * (16 * PPITCH);
    const float LOG2E = 1.44269504f;

    // cooperative async stage of one 32-row K/V tile (8 async ops per wave)
    auto issue_tile = [&](int buf, int j) {
        const int t = threadIdx.x;
        const u32 kb = (u32)(uintptr_t)&kbuf[buf][0];
        const u32 vb = (u32)(uintptr_t)&vbuf[buf][0];
        #pragma unroll
        for (int i = 0; i < 4; ++i) {
            const int c = t + i * 256;
            const int kr = c >> 2, kc = c & 3;      // K: 256 rows x 64B
            async_ld16(kb + (u32)(kr * (KPITCH * 2) + kc * 16),
                       ktb + (size_t)kr * S_ + j + kc * 8);
            const int vr = c >> 5, vc = c & 31;     // V: 32 rows x 512B
            async_ld16(vb + (u32)(vr * (VPITCH * 2) + vc * 16),
                       vbp + (size_t)(j + vr) * E_ + vc * 8);
        }
    };

    issue_tile(0, 0);

    for (int jt = 0; jt < S_ / 32; ++jt) {
        const int cur = jt & 1;
        if (jt + 1 < S_ / 32) {                 // prefetch next tile (overlap)
            issue_tile(cur ^ 1, (jt + 1) * 32);
            wait_async_le8();                   // first 8 (current tile) done
        } else {
            wait_async_0();
        }
        __syncthreads();                        // current tile visible to all waves

        const u16* kbc = &kbuf[cur][0];
        const u16* vbc = &vbuf[cur][0];

        // ---- S = Q * K^T for 32 kv columns (two 16x16 C tiles) ----
        v8f s0 = {}, s1 = {};
        #pragma unroll
        for (int ks = 0; ks < 8; ++ks) {
            BF16x16 a, b0, b1;
            a.u[0] = *(const u32x4*)(qrow + ks * 32 + kh * 8);
            a.u[1] = *(const u32x4*)(qrow + ks * 32 + 16 + kh * 8);
            const u16* kp = kbc + (size_t)(ks * 32) * KPITCH + row;
            #pragma unroll
            for (int s = 0; s < 16; s += 2) {
                const int k = (s & 7) + 8 * kh + (s & 8) * 2;
                b0.u[s >> 3][(s & 7) >> 1] = (u32)kp[k * KPITCH] | ((u32)kp[(k + 1) * KPITCH] << 16);
                b1.u[s >> 3][(s & 7) >> 1] = (u32)kp[k * KPITCH + 16] | ((u32)kp[(k + 1) * KPITCH + 16] << 16);
            }
            s0 = wmma_bf16(a.v, b0.v, s0);
            s1 = wmma_bf16(a.v, b1.v, s1);
        }
        // ---- online softmax (rows live in 16-lane halves per VGPR) ----
        float alpha[8];
        #pragma unroll
        for (int r = 0; r < 8; ++r) {
            float mp = fmaxf(s0[r], s1[r]);
            mp = fmaxf(mp, __shfl_xor(mp, 1));
            mp = fmaxf(mp, __shfl_xor(mp, 2));
            mp = fmaxf(mp, __shfl_xor(mp, 4));
            mp = fmaxf(mp, __shfl_xor(mp, 8));
            const float mn = fmaxf(m[r], mp);
            alpha[r] = exp2f((m[r] - mn) * LOG2E);
            m[r] = mn;
            const float p0 = exp2f((s0[r] - mn) * LOG2E);
            const float p1 = exp2f((s1[r] - mn) * LOG2E);
            s0[r] = p0; s1[r] = p1;
            float sm = p0 + p1;
            sm += __shfl_xor(sm, 1);
            sm += __shfl_xor(sm, 2);
            sm += __shfl_xor(sm, 4);
            sm += __shfl_xor(sm, 8);
            l[r] = l[r] * alpha[r] + sm;
        }
        // ---- P: C-layout -> bf16 in wave-private LDS ----
        {
            const int N = lane & 15;
            #pragma unroll
            for (int r = 0; r < 8; ++r) {
                const int M = r + 8 * kh;
                pw[M * PPITCH + N]      = f2bf(s0[r]);
                pw[M * PPITCH + 16 + N] = f2bf(s1[r]);
            }
        }
        // ---- rescale O ----
        #pragma unroll
        for (int t = 0; t < 16; ++t)
            #pragma unroll
            for (int r = 0; r < 8; ++r) o[t][r] *= alpha[r];
        // ---- reload P in A-matrix layout ----
        BF16x16 pa;
        pa.u[0] = *(const u32x4*)(pw + row * PPITCH + kh * 8);
        pa.u[1] = *(const u32x4*)(pw + row * PPITCH + 16 + kh * 8);
        // ---- O += P * V over 16 f-tiles (V from LDS) ----
        #pragma unroll
        for (int ft = 0; ft < 16; ++ft) {
            const u16* vp = vbc + ft * 16 + row;
            BF16x16 bv;
            #pragma unroll
            for (int s = 0; s < 16; s += 2) {
                const int k = (s & 7) + 8 * kh + (s & 8) * 2;
                bv.u[s >> 3][(s & 7) >> 1] = (u32)vp[k * VPITCH] | ((u32)vp[(k + 1) * VPITCH] << 16);
            }
            o[ft] = wmma_bf16(pa.v, bv.v, o[ft]);
        }
        __syncthreads();    // all waves done reading tile before it is rewritten
    }

    // ---- epilogue: normalize, store per-head O (f32) ----
    float* obase = Obuf + (bh * S_ + q0) * E_;
    const int N = lane & 15;
    #pragma unroll
    for (int r = 0; r < 8; ++r) {
        const float inv = 1.0f / l[r];
        const int M = r + 8 * kh;
        #pragma unroll
        for (int ft = 0; ft < 16; ++ft)
            obase[(size_t)M * E_ + ft * 16 + N] = o[ft][r] * inv;
    }
}

// ------------------------------------------------------------------
// Kernel 4: mean over heads -> d_out
// ------------------------------------------------------------------
__global__ __launch_bounds__(256) void mean_heads_kernel(
    const float* __restrict__ Obuf, float* __restrict__ out)
{
    const size_t SE = (size_t)S_ * E_;
    size_t i = (size_t)blockIdx.x * 256 + threadIdx.x;
    if (i >= (size_t)B_ * SE) return;
    const size_t b = i / SE, rem = i % SE;
    float acc = 0.0f;
    #pragma unroll
    for (int h = 0; h < H_; ++h) acc += Obuf[(b * H_ + h) * SE + rem];
    out[i] = acc * 0.125f;
}

// ------------------------------------------------------------------
extern "C" void kernel_launch(void* const* d_in, const int* in_sizes, int n_in,
                              void* d_out, int out_size, void* d_ws, size_t ws_size,
                              hipStream_t stream) {
    const float* x  = (const float*)d_in[0];
    const float* wq = (const float*)d_in[1];
    const float* wk = (const float*)d_in[2];
    const float* wv = (const float*)d_in[3];
    float* out = (float*)d_out;

    const size_t nX  = (size_t)B_ * S_ * E_;        // 2,097,152
    const size_t nW  = (size_t)H_ * E_ * E_;        //   524,288
    const size_t nQ  = (size_t)B_ * H_ * S_ * E_;   // 16,777,216

    char* ws = (char*)d_ws;
    u16* xb  = (u16*)ws;                 ws += nX * sizeof(u16);
    u16* wqb = (u16*)ws;                 ws += nW * sizeof(u16);
    u16* wkb = (u16*)ws;                 ws += nW * sizeof(u16);
    u16* wvb = (u16*)ws;                 ws += nW * sizeof(u16);
    u16* Qb  = (u16*)ws;                 ws += nQ * sizeof(u16);
    u16* Kt  = (u16*)ws;                 ws += nQ * sizeof(u16);
    u16* Vb  = (u16*)ws;                 ws += nQ * sizeof(u16);
    float* Obuf = (float*)ws;            // nQ floats

    cvt_bf16_kernel<<<(int)(nX / 256), 256, 0, stream>>>(x,  xb,  (int)nX);
    cvt_bf16_kernel<<<(int)(nW / 256), 256, 0, stream>>>(wq, wqb, (int)nW);
    cvt_bf16_kernel<<<(int)(nW / 256), 256, 0, stream>>>(wk, wkb, (int)nW);
    cvt_bf16_kernel<<<(int)(nW / 256), 256, 0, stream>>>(wv, wvb, (int)nW);

    proj_qkv_kernel<<<8192, 256, 0, stream>>>(xb, wqb, wkb, wvb, Qb, Kt, Vb);

    attn_kernel<<<512, 256, 0, stream>>>(Qb, Kt, Vb, Obuf);

    mean_heads_kernel<<<(int)(nX / 256), 256, 0, stream>>>(Obuf, out);
}
